// VCTreePredictor_42365557408495
// MI455X (gfx1250) — matrix operand order repaired and checked by reading, not compiled, not run on hardware
//
#include <hip/hip_runtime.h>

// ---------------- problem constants (from reference) ----------------
#define N_OBJ   5000
#define N_REL   20000
#define HID     512
#define TWOHID  1024
#define POOL    4096
#define NRC     51     // relation classes
#define NOC     151    // object classes
#define NCCPAD  64     // W_cc rows padded to one WMMA tile

// ---------------- types ----------------
typedef __attribute__((ext_vector_type(16))) __bf16        v16bf;
typedef __attribute__((ext_vector_type(2)))  __bf16        bf16x2;
typedef __attribute__((ext_vector_type(8)))  float         v8f;
typedef __attribute__((ext_vector_type(4)))  float         v4f;
typedef __attribute__((ext_vector_type(4)))  unsigned int  v4u;
typedef __attribute__((ext_vector_type(8)))  unsigned int  v8u;
typedef __attribute__((ext_vector_type(2)))  unsigned int  v2u;
typedef __attribute__((ext_vector_type(4)))  int           i32x4;
typedef __attribute__((ext_vector_type(8)))  int           i32x8;

// hardware f32 -> bf16 (RNE)
__device__ __forceinline__ unsigned short f2bf(float f) {
    __bf16 h = (__bf16)f;
    return __builtin_bit_cast(unsigned short, h);
}
__device__ __forceinline__ unsigned int pack2bf(float lo, float hi) {
#if __has_builtin(__builtin_amdgcn_cvt_pk_bf16_f32)
    auto p = __builtin_amdgcn_cvt_pk_bf16_f32(lo, hi);
    return __builtin_bit_cast(unsigned int, p);
#else
    bf16x2 v = { (__bf16)lo, (__bf16)hi };
    return __builtin_bit_cast(unsigned int, v);
#endif
}

// ---------------- CDNA5 async global->LDS copy (ASYNCcnt) ----------------
__device__ __forceinline__ void async_copy16(unsigned lds_off, unsigned long long gaddr) {
    asm volatile("global_load_async_to_lds_b128 %0, %1, off"
                 :: "v"(lds_off), "v"(gaddr)
                 : "memory");
}
__device__ __forceinline__ void wait_async0() {
#if __has_builtin(__builtin_amdgcn_s_wait_asynccnt)
    __builtin_amdgcn_s_wait_asynccnt(0);
#else
    asm volatile("s_wait_asynccnt 0x0" ::: "memory");
#endif
}

// ---------------- CDNA5 Tensor Data Mover: 2D tile -> LDS (TENSORcnt) ----------------
// D# per CDNA5 ISA. data_size=2B, 2D tile.
__device__ __forceinline__ void tdm_load_2d(unsigned lds_off, const void* gptr,
                                            unsigned tile_w, unsigned tile_h,
                                            unsigned tensor_w, unsigned tensor_h,
                                            unsigned row_stride) {
    unsigned long long ga = (unsigned long long)gptr;
    v4u g0;
    g0[0] = 1u;                                       // count=1, user, no gather
    g0[1] = lds_off;                                  // lds_addr
    g0[2] = (unsigned)(ga & 0xFFFFFFFFu);             // global_addr[31:0]
    g0[3] = (unsigned)((ga >> 32) & 0x01FFFFFFu)      // global_addr[56:32]
          | (2u << 30);                               // type = 2 ("image")
    i32x8 g1;
    g1[0] = (int)(1u << 16);                          // data_size=1 (2 bytes), mask=0
    g1[1] = (int)((tensor_w & 0xFFFFu) << 16);        // tensor_dim0[15:0]
    g1[2] = (int)((tensor_w >> 16) & 0xFFFFu)         // tensor_dim0[31:16]
          | (int)((tensor_h & 0xFFFFu) << 16);        // tensor_dim1[15:0]
    g1[3] = (int)((tensor_h >> 16) & 0xFFFFu)         // tensor_dim1[31:16]
          | (int)((tile_w & 0xFFFFu) << 16);          // tile_dim0
    g1[4] = (int)(tile_h & 0xFFFFu);                  // tile_dim1 (tile_dim2=0)
    g1[5] = (int)row_stride;                          // tensor_dim0_stride[31:0]
    g1[6] = 0;
    g1[7] = 0;
    i32x4 z = { 0, 0, 0, 0 };
#if __clang_major__ >= 23
    i32x8 z8 = { 0, 0, 0, 0, 0, 0, 0, 0 };
    __builtin_amdgcn_tensor_load_to_lds(g0, g1, z, z, z8, 0);
#else
    __builtin_amdgcn_tensor_load_to_lds(g0, g1, z, z, 0);
#endif
}
__device__ __forceinline__ void wait_tensor0() {
    __builtin_amdgcn_s_wait_tensorcnt((unsigned short)0);
}

// ---------------- WMMA fragments ----------------
__device__ __forceinline__ v16bf mkfrag(v4u x, v4u y) {
    v8u t;
    t[0]=x[0]; t[1]=x[1]; t[2]=x[2]; t[3]=x[3];
    t[4]=y[0]; t[5]=y[1]; t[6]=y[2]; t[7]=y[3];
    return __builtin_bit_cast(v16bf, t);
}
__device__ __forceinline__ v16bf frag_a(const unsigned short* sh, int mt, int ki,
                                        int ln, int h) {
    const unsigned short* p = sh + (mt + ln) * 64 + ki + 8 * h;
    return mkfrag(*(const v4u*)p, *(const v4u*)(p + 16));
}
__device__ __forceinline__ v16bf frag_b(const unsigned short* sh, int nt, int ki,
                                        int ln, int h) {
    const unsigned short* p = sh + (nt + ln) * 64 + ki + 8 * h;
    return mkfrag(*(const v4u*)p, *(const v4u*)(p + 16));
}
__device__ __forceinline__ v8f wmma_bf16(v16bf a, v16bf b, v8f c) {
    return __builtin_amdgcn_wmma_f32_16x16x32_bf16(
        false, a, false, b, (short)0, c, false, false);
}

__device__ __forceinline__ int imin(int a, int b) { return a < b ? a : b; }

// ---------------- kernel 0: weight precision prep ----------------
__global__ __launch_bounds__(256) void k_prep(const float* __restrict__ W_pc,
                                              const float* __restrict__ W_cc,
                                              unsigned int* __restrict__ Wpc_bf,
                                              unsigned int* __restrict__ Wcc_bf) {
    const int NPC2 = (POOL * TWOHID) / 2;
    const int NCC2 = (NCCPAD * POOL) / 2;
    int idx = blockIdx.x * 256 + threadIdx.x;
    if (idx < NPC2) {
        Wpc_bf[idx] = pack2bf(W_pc[2 * idx], W_pc[2 * idx + 1]);
    } else if (idx < NPC2 + NCC2) {
        int k = idx - NPC2;
        int c = k >> 11;
        int col2 = k & 2047;
        Wcc_bf[k] = (c < NRC)
            ? pack2bf(W_cc[c * POOL + 2 * col2], W_cc[c * POOL + 2 * col2 + 1])
            : 0u;
    }
}

// ---------------- kernel 1: edge_rep = relu(edge_ctx @ W_pe^T + b_pe) -> bf16 ----------------
// Out-of-range rows are index-clamped (not zero-filled): their outputs are
// store-masked, so values are don't-care -> no divergent guards in hot loops.
__global__ __launch_bounds__(256) void k_gemm1(const float* __restrict__ edge_ctx,
                                               const float* __restrict__ W_pe,
                                               const float* __restrict__ b_pe,
                                               unsigned short* __restrict__ edge_bf) {
    __shared__ unsigned short As[64 * 64];
    __shared__ unsigned short Bs[64 * 64];

    const int t = threadIdx.x;
    const int lane = t & 31, wave = t >> 5;
    const int ln = lane & 15, h = lane >> 4;
    const int mbase = blockIdx.x * 64;
    const int nbase = blockIdx.y * 64;
    const bool fullm = (mbase + 64 <= N_OBJ);

    const int tid0 = 2 * wave, tid1 = tid0 + 1;
    const int tm0 = (tid0 >> 2) * 16, tn0 = (tid0 & 3) * 16;
    const int tm1 = (tid1 >> 2) * 16, tn1 = (tid1 & 3) * 16;

    v8f acc0 = {}; v8f acc1 = {};

    for (int kb = 0; kb < HID; kb += 64) {
        __syncthreads();
        for (int j = 0; j < 4; ++j) {
            int q = t + j * 256;
            int row = q >> 4, col4 = (q & 15) * 4;
            int mg = imin(mbase + row, N_OBJ - 1);           // clamp, no branch
            v4f a = *(const v4f*)&edge_ctx[mg * HID + kb + col4];
            v2u ap = { pack2bf(a[0], a[1]), pack2bf(a[2], a[3]) };
            *(v2u*)&As[row * 64 + col4] = ap;
            v4f b = *(const v4f*)&W_pe[(nbase + row) * HID + kb + col4];
            v2u bp = { pack2bf(b[0], b[1]), pack2bf(b[2], b[3]) };
            *(v2u*)&Bs[row * 64 + col4] = bp;
        }
        __syncthreads();
        for (int ki = 0; ki < 64; ki += 32) {
            acc0 = wmma_bf16(frag_a(As, tm0, ki, ln, h), frag_b(Bs, tn0, ki, ln, h), acc0);
            acc1 = wmma_bf16(frag_a(As, tm1, ki, ln, h), frag_b(Bs, tn1, ki, ln, h), acc1);
        }
    }

    for (int r = 0; r < 8; ++r) {
        int m0 = mbase + tm0 + 8 * h + r;
        int n0 = nbase + tn0 + ln;
        float v0 = acc0[r] + b_pe[n0];
        if (fullm || m0 < N_OBJ) edge_bf[m0 * TWOHID + n0] = f2bf(v0 > 0.0f ? v0 : 0.0f);
        int m1 = mbase + tm1 + 8 * h + r;
        int n1 = nbase + tn1 + ln;
        float v1 = acc1[r] + b_pe[n1];
        if (fullm || m1 < N_OBJ) edge_bf[m1 * TWOHID + n1] = f2bf(v1 > 0.0f ? v1 : 0.0f);
    }
}

// ---------------- kernel 2: fused gather + GEMM2 + (*union + b_pc) + GEMM3 + freq ----------------
__global__ __launch_bounds__(256) void k_rel(const unsigned short* __restrict__ edge_bf,
                                             const unsigned short* __restrict__ Wpc_bf,
                                             const unsigned short* __restrict__ Wcc_bf,
                                             const float* __restrict__ union_f,
                                             const float* __restrict__ b_pc,
                                             const float* __restrict__ b_cc,
                                             const float* __restrict__ freq_emb,
                                             const int* __restrict__ obj_preds,
                                             const int* __restrict__ pair_idx,
                                             float* __restrict__ out) {
    __shared__ unsigned short As[2][64 * 64];  // gathered prod tile (double buffer)
    __shared__ unsigned short Bs[2][64 * 64];  // Wpc chunk (TDM, double buffer)
    __shared__ unsigned short Ss[64 * 64];     // scaled product = A of GEMM3
    __shared__ unsigned short Cs[64 * 64];     // Wcc chunk (TDM prefetch)
    __shared__ int sh_hi[64], sh_ti[64], sh_lbl[64];

    const int t = threadIdx.x;
    const int lane = t & 31, wave = t >> 5;
    const int ln = lane & 15, h = lane >> 4;
    const int mbase = blockIdx.x * 64;
    const bool fullm = (mbase + 64 <= N_REL);

    if (t < 64) {
        int mg = imin(mbase + t, N_REL - 1);   // clamp: masked rows are don't-care
        int hi = pair_idx[2 * mg];
        int ti = pair_idx[2 * mg + 1];
        sh_hi[t] = hi; sh_ti[t] = ti;
        sh_lbl[t] = obj_preds[hi] * NOC + obj_preds[ti];
    }

    const int tid0 = 2 * wave, tid1 = tid0 + 1;
    const int tm0 = (tid0 >> 2) * 16, tn0 = (tid0 & 3) * 16;
    const int tm1 = (tid1 >> 2) * 16, tn1 = (tid1 & 3) * 16;

    // this thread's two 16B A-gather chunks
    const int c0 = t * 2, c1 = c0 + 1;
    const int row0 = c0 >> 3, col0 = (c0 & 7) * 8;
    const int row1 = c1 >> 3, col1 = (c1 & 7) * 8;

    v8f acc3_0 = {}; v8f acc3_1 = {};
    __syncthreads();

    // hoist gather base byte-addresses (fixed for the whole kernel)
    const unsigned long long aH0 =
        (unsigned long long)&edge_bf[(unsigned)sh_hi[row0] * TWOHID + col0];
    const unsigned long long aT0 =
        (unsigned long long)&edge_bf[(unsigned)sh_ti[row0] * TWOHID + col0];
    const unsigned long long aH1 =
        (unsigned long long)&edge_bf[(unsigned)sh_hi[row1] * TWOHID + col1];
    const unsigned long long aT1 =
        (unsigned long long)&edge_bf[(unsigned)sh_ti[row1] * TWOHID + col1];
    const unsigned ldsA0_0 = (unsigned)(unsigned long long)&As[0][row0 * 64 + col0];
    const unsigned ldsA0_1 = (unsigned)(unsigned long long)&As[0][row1 * 64 + col1];
    const unsigned ldsA1_0 = (unsigned)(unsigned long long)&As[1][row0 * 64 + col0];
    const unsigned ldsA1_1 = (unsigned)(unsigned long long)&As[1][row1 * 64 + col1];

    for (int nb = 0; nb < POOL; nb += 64) {
        __syncthreads();   // all waves done with As/Bs/Cs of previous chunk

        if (wave == 0) {
            // GEMM3 B chunk: completes before the K loop (tensorcnt wait)
            tdm_load_2d((unsigned)(unsigned long long)&Cs[0], &Wcc_bf[nb],
                        64, 64, POOL, NCCPAD, POOL);
            // GEMM2 B chunk kb=0
            tdm_load_2d((unsigned)(unsigned long long)&Bs[0][0],
                        &Wpc_bf[(unsigned)nb * TWOHID],
                        64, 64, TWOHID, POOL, TWOHID);
        }
        // A chunk kb=0 (head gather), buffer 0
        async_copy16(ldsA0_0, aH0);
        async_copy16(ldsA0_1, aH1);
        wait_async0();
        if (wave == 0) wait_tensor0();
        __syncthreads();

        v8f acc2_0 = {}; v8f acc2_1 = {};
        int ib = 0;
        for (int kb = 0; kb < TWOHID; kb += 64, ib ^= 1) {
            int kn = kb + 64;
            if (kn < TWOHID) {   // stage next chunk into other buffer
                if (wave == 0)
                    tdm_load_2d((unsigned)(unsigned long long)&Bs[ib ^ 1][0],
                                &Wpc_bf[(unsigned)nb * TWOHID + kn],
                                64, 64, TWOHID, POOL, TWOHID);
                bool head = (kn < HID);                       // uniform select
                unsigned long long g0a = (head ? aH0 : aT0) + 2ull * kn;
                unsigned long long g1a = (head ? aH1 : aT1) + 2ull * kn;
                async_copy16(ib ? ldsA0_0 : ldsA1_0, g0a);
                async_copy16(ib ? ldsA0_1 : ldsA1_1, g1a);
            }
            // compute on current buffer while next streams in
            for (int ki = 0; ki < 64; ki += 32) {
                acc2_0 = wmma_bf16(frag_a(As[ib], tm0, ki, ln, h),
                                   frag_b(Bs[ib], tn0, ki, ln, h), acc2_0);
                acc2_1 = wmma_bf16(frag_a(As[ib], tm1, ki, ln, h),
                                   frag_b(Bs[ib], tn1, ki, ln, h), acc2_1);
            }
            wait_async0();
            if (wave == 0) wait_tensor0();
            __syncthreads();
        }

        // S = (prod_rep + b_pc) * union -> bf16 in Ss; row index clamped (no branches)
        {
            int nl0 = tn0 + ln, nl1 = tn1 + ln;
            float bp0 = b_pc[nb + nl0], bp1 = b_pc[nb + nl1];
            for (int r = 0; r < 8; ++r) {
                int ml0 = tm0 + 8 * h + r;
                long mg0 = imin(mbase + ml0, N_REL - 1);
                float s0 = (acc2_0[r] + bp0) *
                    __builtin_nontemporal_load(&union_f[mg0 * POOL + nb + nl0]);
                Ss[ml0 * 64 + nl0] = f2bf(s0);
                int ml1 = tm1 + 8 * h + r;
                long mg1 = imin(mbase + ml1, N_REL - 1);
                float s1 = (acc2_1[r] + bp1) *
                    __builtin_nontemporal_load(&union_f[mg1 * POOL + nb + nl1]);
                Ss[ml1 * 64 + nl1] = f2bf(s1);
            }
        }
        __syncthreads();   // Ss visible; Cs arrived long ago (tensorcnt waited)

        // GEMM3 partial: acc3 += Ss @ Wcc_pad^T
        for (int ki = 0; ki < 64; ki += 32) {
            acc3_0 = wmma_bf16(frag_a(Ss, tm0, ki, ln, h), frag_b(Cs, tn0, ki, ln, h), acc3_0);
            acc3_1 = wmma_bf16(frag_a(Ss, tm1, ki, ln, h), frag_b(Cs, tn1, ki, ln, h), acc3_1);
        }
    }

    // epilogue: + b_cc + freq_emb[pair_lbl], NT store fp32
    {
        int n0 = tn0 + ln, n1 = tn1 + ln;
        if (n0 < NRC) {
            for (int r = 0; r < 8; ++r) {
                int ml = tm0 + 8 * h + r, mg = mbase + ml;
                float v = acc3_0[r] + b_cc[n0] + freq_emb[sh_lbl[ml] * NRC + n0];
                if (fullm || mg < N_REL)
                    __builtin_nontemporal_store(v, &out[mg * NRC + n0]);
            }
        }
        if (n1 < NRC) {
            for (int r = 0; r < 8; ++r) {
                int ml = tm1 + 8 * h + r, mg = mbase + ml;
                float v = acc3_1[r] + b_cc[n1] + freq_emb[sh_lbl[ml] * NRC + n1];
                if (fullm || mg < N_REL)
                    __builtin_nontemporal_store(v, &out[mg * NRC + n1]);
            }
        }
    }
}

// ---------------- host launcher ----------------
extern "C" void kernel_launch(void* const* d_in, const int* in_sizes, int n_in,
                              void* d_out, int out_size, void* d_ws, size_t ws_size,
                              hipStream_t stream) {
    (void)in_sizes; (void)n_in; (void)out_size; (void)ws_size;

    const float* edge_ctx = (const float*)d_in[0];
    const float* union_f  = (const float*)d_in[1];
    const float* W_pe     = (const float*)d_in[2];
    const float* b_pe     = (const float*)d_in[3];
    const float* W_pc     = (const float*)d_in[4];
    const float* b_pc     = (const float*)d_in[5];
    const float* W_cc     = (const float*)d_in[6];
    const float* b_cc     = (const float*)d_in[7];
    const float* freq_emb = (const float*)d_in[8];
    const int*   obj_preds= (const int*)d_in[9];
    const int*   pair_idx = (const int*)d_in[10];
    float* out = (float*)d_out;

    char* ws = (char*)d_ws;
    unsigned short* edge_bf = (unsigned short*)ws;                        // 10,240,000 B
    unsigned short* Wpc_bf  = (unsigned short*)(ws + 10240000);           //  8,388,608 B
    unsigned short* Wcc_bf  = (unsigned short*)(ws + 10240000 + 8388608); //    524,288 B

    k_prep<<<8704, 256, 0, stream>>>(W_pc, W_cc,
                                     (unsigned int*)Wpc_bf, (unsigned int*)Wcc_bf);
    k_gemm1<<<dim3(79, 16), 256, 0, stream>>>(edge_ctx, W_pe, b_pe, edge_bf);
    k_rel<<<313, 256, 0, stream>>>(edge_bf, Wpc_bf, Wcc_bf, union_f,
                                   b_pc, b_cc, freq_emb, obj_preds, pair_idx, out);
}
